// _DeCompressJPEG_42992622633231
// MI455X (gfx1250) — matrix operand order repaired and checked by reading, not compiled, not run on hardware
//
#include <hip/hip_runtime.h>

typedef __attribute__((ext_vector_type(16))) _Float16 v16h;
typedef __attribute__((ext_vector_type(8)))  float    v8f;

#define HDIM 2048
#define WDIM 2048

// One workgroup = one 32x32 RGB region of one batch image.
// Waves 0-3: four 16x16 luma tiles (each = 2x2 Y blocks) via WMMA IDCT.
// Wave 4: 16x16 Cb tile; Wave 5: 16x16 Cr tile. Waves 6-7 idle in phase 1.
// Phase 2: all 256 threads do chroma upsample + YCbCr->RGB + clip + store.
__global__ __launch_bounds__(256)
void jpeg_decode_fused(const float* __restrict__ ycoef,
                       const float* __restrict__ cbcoef,
                       const float* __restrict__ crcoef,
                       const float* __restrict__ y_table,
                       const float* __restrict__ c_table,
                       const float* __restrict__ alpha,
                       const float* __restrict__ dct,     // [8][8][8][8]; C[x][u] = dct[x*512 + u*8]
                       const float* __restrict__ shift,   // [3]
                       const float* __restrict__ matrix,  // [3][3] (c,d)
                       const int*   __restrict__ factor_p,
                       float* __restrict__ out)           // [B,3,H,W]
{
    __shared__ float Cm[64];          // C[x][u] cosine matrix (from dct_tensor[x,0,u,0])
    __shared__ float scl[2][64];      // premultiplied dequant: [0]=Y, [1]=chroma
    __shared__ float yT[32][33];      // decoded luma region (padded)
    __shared__ float cbT[16][17];     // decoded chroma tiles
    __shared__ float crT[16][17];
    __shared__ float Us[6][16][17];   // per-wave stage-1 intermediate U = X16 @ C16

    const int tid  = threadIdx.x;
    const int wave = tid >> 5;
    const int lane = tid & 31;
    const int rw = blockIdx.x;        // region col (32 px units)
    const int rh = blockIdx.y;        // region row
    const int bb = blockIdx.z;        // batch

    const float factorf = (float)factor_p[0];

    // ---- stage small constant tables into LDS (no divergent global loads later) ----
    if (tid < 64) {
        Cm[tid] = dct[(size_t)(tid >> 3) * 512 + (size_t)(tid & 7) * 8];
        const float a = alpha[tid] * factorf;
        scl[0][tid] = y_table[tid] * a;
        scl[1][tid] = c_table[tid] * a;
    }
    __syncthreads();

    if (wave < 6) {
        // ---- plane / tile selection ----
        const float* coeff;
        int sel, blocksPerRow, nBlocks, tr, tc, bh0, bw0;
        if (wave < 4) {
            coeff = ycoef; sel = 0;
            blocksPerRow = WDIM / 8; nBlocks = (HDIM / 8) * (WDIM / 8);
            tr = wave >> 1; tc = wave & 1;
            bh0 = rh * 4 + tr * 2; bw0 = rw * 4 + tc * 2;
        } else {
            coeff = (wave == 4) ? cbcoef : crcoef; sel = 1;
            blocksPerRow = WDIM / 16; nBlocks = (HDIM / 16) * (WDIM / 16);
            tr = 0; tc = 0;
            bh0 = rh * 2; bw0 = rw * 2;
        }

        const int M    = lane & 15;   // tile row (A-row)
        const int half = lane >> 4;   // lanes<16: K=0..7 ; lanes>=16: K=8..15
        const int qr   = M >> 3;      // block row within 16x16 tile
        const int xr   = M & 7;       // row within 8x8 block
        // A covers K = 8*half .. 8*half+7 -> block column qc = half
        const int nblk = (bh0 + qr) * blocksPerRow + (bw0 + half);
        const float* src = coeff + ((size_t)bb * nBlocks + nblk) * 64 + (size_t)xr * 8;
        const float* sc  = &scl[sel][xr * 8];

        // ---- A1 = dequantized coefficient tile X16 (16x32 f16, K>=16 zero) ----
        v16h a1;
        #pragma unroll
        for (int e = 0; e < 8; ++e)
            a1[e] = (_Float16)(src[e] * sc[e]);
        #pragma unroll
        for (int e = 8; e < 16; ++e) a1[e] = (_Float16)0.0f;

        // ---- B1 = C16 = diag(C, C)  (32x16 f16; element e<8 of half h is K=8h+e, e>=8 is K>=16 -> zero) ----
        // B1[K][N] = C[K%8][N%8] iff K/8 == N/8 ; N = lane & 15
        const int n8 = lane & 7;
        v16h b1;
        #pragma unroll
        for (int e = 0; e < 16; ++e) {
            const float v = Cm[(e & 7) * 8 + n8];                 // unconditional LDS read
            const bool on = (e < 8) && (half == ((lane >> 3) & 1));
            b1[e] = on ? (_Float16)v : (_Float16)0.0f;            // v_cndmask, no exec branch
        }

        v8f czero = {};
        // U = X16 @ C16   (each quadrant: Xblk @ C)
        v8f u = __builtin_amdgcn_wmma_f32_16x16x32_f16(
            false, a1, false, b1, (short)0, czero, false, false);

        // scatter U (C/D layout) to row-major LDS scratch
        #pragma unroll
        for (int r = 0; r < 8; ++r)
            Us[wave][r + 8 * half][M] = u[r];

        // ---- A2 = Ct16 = diag(C^T, C^T) (16x32 f16) ----
        // row M, this lane covers K = 8*half..+7; Ct16[M][K] = C[K%8][M%8] iff K/8==M/8
        v16h a2;
        #pragma unroll
        for (int e = 0; e < 8; ++e) {
            const float v = Cm[e * 8 + xr];                       // unconditional LDS read
            a2[e] = (qr == half) ? (_Float16)v : (_Float16)0.0f;
        }
        #pragma unroll
        for (int e = 8; e < 16; ++e) a2[e] = (_Float16)0.0f;

        // ---- B2 = U (32x16 f16; element e<8 of half h holds K=8h+e, e>=8 zero) ----
        v16h b2;
        #pragma unroll
        for (int e = 0; e < 16; ++e) {
            const float uv = Us[wave][(e & 7) + 8 * half][lane & 15]; // unconditional LDS read
            b2[e] = (e < 8) ? (_Float16)uv : (_Float16)0.0f;
        }

        // Out = Ct16 @ U  -> four decoded 8x8 IDCT blocks in one 16x16 tile
        v8f o = __builtin_amdgcn_wmma_f32_16x16x32_f16(
            false, a2, false, b2, (short)0, czero, false, false);

        #pragma unroll
        for (int r = 0; r < 8; ++r) {
            const float px = 0.25f * o[r] + 128.0f;
            const int rr = r + 8 * half;          // tile row of this element
            if (wave < 4)       yT[tr * 16 + rr][tc * 16 + M] = px;
            else if (wave == 4) cbT[rr][M] = px;
            else                crT[rr][M] = px;
        }
    }

    __syncthreads();

    // ---- phase 2: upsample + color convert + clip + store (coalesced) ----
    const float s0 = shift[0], s1 = shift[1], s2 = shift[2];
    float m[9];
    #pragma unroll
    for (int i = 0; i < 9; ++i) m[i] = matrix[i];

    const int px = tid & 31;
    #pragma unroll
    for (int k = 0; k < 4; ++k) {
        const int py = (tid >> 5) + 8 * k;
        const float c0 = yT[py][px]            + s0;
        const float c1 = cbT[py >> 1][px >> 1] + s1;
        const float c2 = crT[py >> 1][px >> 1] + s2;
        #pragma unroll
        for (int d = 0; d < 3; ++d) {
            float v = c0 * m[d] + c1 * m[3 + d] + c2 * m[6 + d];
            v = fminf(fmaxf(v, 0.0f), 255.0f) * (1.0f / 255.0f);
            const size_t oidx = (((size_t)bb * 3 + d) * HDIM + (size_t)(rh * 32 + py)) * WDIM
                                + (size_t)(rw * 32 + px);
            out[oidx] = v;
        }
    }
}

extern "C" void kernel_launch(void* const* d_in, const int* in_sizes, int n_in,
                              void* d_out, int out_size, void* d_ws, size_t ws_size,
                              hipStream_t stream) {
    const float* y   = (const float*)d_in[0];
    const float* cb  = (const float*)d_in[1];
    const float* cr  = (const float*)d_in[2];
    const float* yt  = (const float*)d_in[3];
    const float* ct  = (const float*)d_in[4];
    const float* al  = (const float*)d_in[5];
    const float* dct = (const float*)d_in[6];
    const float* sh  = (const float*)d_in[7];
    const float* mx  = (const float*)d_in[8];
    const int*   fac = (const int*)d_in[11];

    const int B = in_sizes[0] / (HDIM * WDIM);   // y has B*H*W elements

    dim3 grid(WDIM / 32, HDIM / 32, B);          // one block per 32x32 region
    jpeg_decode_fused<<<grid, 256, 0, stream>>>(
        y, cb, cr, yt, ct, al, dct, sh, mx, fac, (float*)d_out);
}
